// GNN_39316130627646
// MI455X (gfx1250) — matrix operand order repaired and checked by reading, compiled, not run on hardware
//
#include <hip/hip_runtime.h>

// ---------------------------------------------------------------------------
// GNN forward for gfx1250 (MI455X). All GEMMs: v_wmma_f32_16x16x32_bf16.
// - node features + weights in bf16 (halves dominant HBM traffic)
// - weights pre-swizzled into WMMA B-fragment order (2x b128 per fragment)
// - edge |a-b| via native v_pk_add_bf16 (inline asm, neg modifier)
// - node GEMM A-staging: GLOBAL_LOAD_ASYNC_TO_LDS_B128 + double buffer
//   (1 barrier per K-tile) when the builtin is available; sync fallback else.
// N=50000, E=400000, IN=64, HID=256, OUT=128.
// ---------------------------------------------------------------------------

#define GNN_N   50000
#define GNN_E   400000
#define GNN_IN  64
#define GNN_HID 256
#define GNN_OUT 128
#define GNN_B   64

typedef __attribute__((ext_vector_type(16))) __bf16 v16bf;
typedef __attribute__((ext_vector_type(8)))  __bf16 v8bf;
typedef __attribute__((ext_vector_type(8)))  float  v8f;
typedef int v4i __attribute__((vector_size(16)));   // matches builtin prototype

#define A_LDSW 72   // LDS row stride (bf16 elems) for the 32x64 A tile (pad 8)

// epilogue flags (compile-time template params)
#define F_RELU  1
#define F_ACC   2
#define F_BIAS  4
#define F_OUTBF 8

#if defined(__has_builtin)
#  if __has_builtin(__builtin_amdgcn_global_load_async_to_lds_b128)
#    define HAVE_ASYNC_LDS 1
#  endif
#endif
#ifndef HAVE_ASYNC_LDS
#  define HAVE_ASYNC_LDS 0
#endif

#define AS1 __attribute__((address_space(1)))
#define AS3 __attribute__((address_space(3)))

__device__ __forceinline__ void wait_asynccnt0() {
    asm volatile("s_wait_asynccnt 0" ::: "memory");
}

__device__ __forceinline__ __bf16 f2bf(float f) {
    unsigned u = __builtin_bit_cast(unsigned, f);
    u += 0x7FFFu + ((u >> 16) & 1u);          // round-to-nearest-even
    unsigned short h = (unsigned short)(u >> 16);
    return __builtin_bit_cast(__bf16, h);
}
__device__ __forceinline__ float bf2f(__bf16 h) {
    unsigned short s = __builtin_bit_cast(unsigned short, h);
    unsigned u = ((unsigned)s) << 16;
    return __builtin_bit_cast(float, u);
}
__device__ __forceinline__ float bf2f_u16(unsigned short s) {
    unsigned u = ((unsigned)s) << 16;
    return __builtin_bit_cast(float, u);
}

// packed |a-b| on 2x bf16 via native VOP3P packed bf16 add with negate,
// then clear sign bits. (CDNA5 V_PK_ADD_BF16, ISA 15.10 opcode 35)
__device__ __forceinline__ unsigned absdiff_pk_bf16(unsigned a, unsigned b) {
    unsigned d;
    asm("v_pk_add_bf16 %0, %1, %2 neg_lo:[0,1] neg_hi:[0,1]"
        : "=v"(d) : "v"(a), "v"(b));
    return d & 0x7FFF7FFFu;
}

// A fragment (16x32 bf16, MxK) per ISA layout, from LDS tile at column colOff:
// lane = half*16 + m ; elems e=0..7 -> K = half*8+e ; e=8..15 -> K = 16+half*8+(e-8)
__device__ __forceinline__ v16bf load_a_frag(const __bf16* sA, int colOff, int lane) {
    int m = lane & 15, half = lane >> 4;
    const __bf16* p = sA + m * A_LDSW + colOff + half * 8;
    v8bf lo = *(const v8bf*)(p);        // K = half*8 .. +7
    v8bf hi = *(const v8bf*)(p + 16);   // K = 16+half*8 .. +7
    v16bf a;
#pragma unroll
    for (int i = 0; i < 8; ++i) { a[i] = lo[i]; a[i + 8] = hi[i]; }
    return a;
}

// B fragment from pre-swizzled weights Wp[K/32][Ncols][32]:
// lane = half*16 + n ; elems 0..7 -> K = half*8.., 8..15 -> K = 16+half*8.. -> two b128 loads.
__device__ __forceinline__ v16bf load_b_packed(const __bf16* __restrict__ Wp, int Ncols,
                                               int kk, int n0, int lane) {
    const __bf16* bp = Wp + ((size_t)(kk >> 5) * Ncols + n0 + (lane & 15)) * 32
                          + (lane >> 4) * 8;
    v8bf lo = *(const v8bf*)(bp);
    v8bf hi = *(const v8bf*)(bp + 16);
    v16bf b;
#pragma unroll
    for (int i = 0; i < 8; ++i) { b[i] = lo[i]; b[i + 8] = hi[i]; }
    return b;
}

// ---------------------------------------------------------------------------
// Node GEMM: C[M,Ncols] = act( A[M,K] @ W[K,Ncols] + bias [+ Cin] )
// Block: 256 thr = 8 waves; tile 32 rows x 128 cols; K-step 64 (4 WMMA/wave).
// grid = ((M+31)/32, Ncols/128). A source fp32 (AF32) or bf16.
// bf16 path: async-to-LDS staging with double buffer (1 barrier / K-tile).
// ---------------------------------------------------------------------------
template<bool AF32, int FLAGS>
__global__ void __launch_bounds__(256)
node_gemm(const void* __restrict__ Asrc, int lda,
          const __bf16* __restrict__ Wp, int Ncols,
          const float* __restrict__ bias,
          const float* __restrict__ Cin, int ldcin,
          float* __restrict__ Cf,
          __bf16* __restrict__ Cb,
          int ldc, int M, int K)
{
    __shared__ __align__(16) __bf16 sA[2][32 * A_LDSW];
    const int tid  = threadIdx.x;
    const int wave = tid >> 5, lane = tid & 31;
    const int m0 = blockIdx.x * 32;
    const int n0 = blockIdx.y * 128 + wave * 16;

    // staging: each thread copies 8 contiguous K elems of one row (b128)
    const int sr = tid >> 3;            // 0..31
    const int sk = (tid & 7) * 8;       // 0,8,..,56
    const int srcRow = min(m0 + sr, M - 1);

    v8f acc0 = {}, acc1 = {};

#if HAVE_ASYNC_LDS
    constexpr bool USE_ASYNC = !AF32;
#else
    constexpr bool USE_ASYNC = false;
#endif

    if constexpr (USE_ASYNC) {
#if HAVE_ASYNC_LDS
        const __bf16* abase = (const __bf16*)Asrc + (size_t)srcRow * lda + sk;
        // prologue: stage tile 0 into buffer 0
        __builtin_amdgcn_global_load_async_to_lds_b128(
            (AS1 v4i*)const_cast<__bf16*>(abase),
            (AS3 v4i*)&sA[0][sr * A_LDSW + sk], 0, 0);
        wait_asynccnt0();
        __syncthreads();
        int cur = 0;
        for (int kk = 0; kk < K; kk += 64) {
            if (kk + 64 < K) {  // overlap: stage next tile into other buffer
                __builtin_amdgcn_global_load_async_to_lds_b128(
                    (AS1 v4i*)const_cast<__bf16*>(abase + kk + 64),
                    (AS3 v4i*)&sA[cur ^ 1][sr * A_LDSW + sk], 0, 0);
                __builtin_prefetch(Wp + ((size_t)((kk + 64) >> 5) * Ncols + n0 + (lane & 15)) * 32, 0, 3);
            }
            const __bf16* buf = sA[cur];
#pragma unroll
            for (int s = 0; s < 2; ++s) {
                v16bf a0 = load_a_frag(buf, s * 32, lane);
                v16bf a1 = load_a_frag(buf + 16 * A_LDSW, s * 32, lane);
                v16bf b  = load_b_packed(Wp, Ncols, kk + s * 32, n0, lane);
                acc0 = __builtin_amdgcn_wmma_f32_16x16x32_bf16(false, a0, false, b, (short)0, acc0, false, false);
                acc1 = __builtin_amdgcn_wmma_f32_16x16x32_bf16(false, a1, false, b, (short)0, acc1, false, false);
            }
            wait_asynccnt0();
            __syncthreads();
            cur ^= 1;
        }
#endif
    } else {
        for (int kk = 0; kk < K; kk += 64) {
            if constexpr (AF32) {
                const float* ap = (const float*)Asrc + (size_t)srcRow * lda + kk + sk;
                float4 v0 = ((const float4*)ap)[0];
                float4 v1 = ((const float4*)ap)[1];
                __bf16* dp = &sA[0][sr * A_LDSW + sk];
                dp[0] = f2bf(v0.x); dp[1] = f2bf(v0.y); dp[2] = f2bf(v0.z); dp[3] = f2bf(v0.w);
                dp[4] = f2bf(v1.x); dp[5] = f2bf(v1.y); dp[6] = f2bf(v1.z); dp[7] = f2bf(v1.w);
            } else {
                const __bf16* ap = (const __bf16*)Asrc + (size_t)srcRow * lda + kk + sk;
                *(uint4*)&sA[0][sr * A_LDSW + sk] = *(const uint4*)ap;
            }
            if (kk + 64 < K)
                __builtin_prefetch(Wp + ((size_t)((kk + 64) >> 5) * Ncols + n0 + (lane & 15)) * 32, 0, 3);
            __syncthreads();
#pragma unroll
            for (int s = 0; s < 2; ++s) {
                v16bf a0 = load_a_frag(sA[0], s * 32, lane);
                v16bf a1 = load_a_frag(sA[0] + 16 * A_LDSW, s * 32, lane);
                v16bf b  = load_b_packed(Wp, Ncols, kk + s * 32, n0, lane);
                acc0 = __builtin_amdgcn_wmma_f32_16x16x32_bf16(false, a0, false, b, (short)0, acc0, false, false);
                acc1 = __builtin_amdgcn_wmma_f32_16x16x32_bf16(false, a1, false, b, (short)0, acc1, false, false);
            }
            __syncthreads();
        }
    }

    // epilogue: lane holds (M = t*16 + r + 8*(lane>>4), N = lane&15)
    const int n = lane & 15, hsel = lane >> 4;
    float bv = 0.0f;
    if constexpr (FLAGS & F_BIAS) bv = bias[n0 + n];
#pragma unroll
    for (int t = 0; t < 2; ++t) {
        v8f& acc = t ? acc1 : acc0;
#pragma unroll
        for (int r = 0; r < 8; ++r) {
            int m = m0 + t * 16 + r + 8 * hsel;
            if (m < M) {
                float v = acc[r] + bv;
                if constexpr (FLAGS & F_ACC)  v += Cin[(size_t)m * ldcin + n0 + n];
                if constexpr (FLAGS & F_RELU) v = fmaxf(v, 0.0f);
                if constexpr (FLAGS & F_OUTBF) Cb[(size_t)m * ldc + n0 + n] = f2bf(v);
                else                           Cf[(size_t)m * ldc + n0 + n] = v;
            }
        }
    }
}

// ---------------------------------------------------------------------------
// Edge GEMM fused: h = relu(|x[row]-x[col]| @ W1 + b1) ;
// wsum[e] += sum_n h[e,n] * W2[n]  (second linear fused; no [E,256] buffer).
// Block tile 32 edges x 128 cols ; grid = (E/32, HID/128). X is bf16.
// ---------------------------------------------------------------------------
__global__ void __launch_bounds__(256)
edge_gemm(const __bf16* __restrict__ X, int ldx,
          const int* __restrict__ erow, const int* __restrict__ ecol,
          const __bf16* __restrict__ Wp, int Ncols,
          const float* __restrict__ b1,
          const float* __restrict__ W2,
          float* __restrict__ wsum, int K)
{
    __shared__ __align__(16) __bf16 sA[32 * A_LDSW];
    const int tid  = threadIdx.x;
    const int wave = tid >> 5, lane = tid & 31;
    const int e0 = blockIdx.x * 32;
    const int n0 = blockIdx.y * 128 + wave * 16;

    const int sr = tid >> 3;            // edge row in tile, 0..31
    const int sk = (tid & 7) * 8;
    const int e  = e0 + sr;             // E % 32 == 0, always valid
    const size_t rs = (size_t)erow[e] * ldx;
    const size_t cs = (size_t)ecol[e] * ldx;

    v8f acc0 = {}, acc1 = {};
    for (int kk = 0; kk < K; kk += 64) {
        // gather 8 bf16 per side, |a-b| in packed bf16 (v_pk_add_bf16 + mask)
        uint4 us = *(const uint4*)(const void*)(X + rs + kk + sk);
        uint4 ud = *(const uint4*)(const void*)(X + cs + kk + sk);
        uint4 o;
        o.x = absdiff_pk_bf16(us.x, ud.x);
        o.y = absdiff_pk_bf16(us.y, ud.y);
        o.z = absdiff_pk_bf16(us.z, ud.z);
        o.w = absdiff_pk_bf16(us.w, ud.w);
        *(uint4*)&sA[sr * A_LDSW + sk] = o;
        if (kk + 64 < K) {
            __builtin_prefetch(X + rs + kk + 64 + sk, 0, 0);
            __builtin_prefetch(X + cs + kk + 64 + sk, 0, 0);
            __builtin_prefetch(Wp + ((size_t)((kk + 64) >> 5) * Ncols + n0 + (lane & 15)) * 32, 0, 3);
        }
        __syncthreads();
#pragma unroll
        for (int s = 0; s < 2; ++s) {
            v16bf a0 = load_a_frag(sA, s * 32, lane);
            v16bf a1 = load_a_frag(sA + 16 * A_LDSW, s * 32, lane);
            v16bf b  = load_b_packed(Wp, Ncols, kk + s * 32, n0, lane);
            acc0 = __builtin_amdgcn_wmma_f32_16x16x32_bf16(false, a0, false, b, (short)0, acc0, false, false);
            acc1 = __builtin_amdgcn_wmma_f32_16x16x32_bf16(false, a1, false, b, (short)0, acc1, false, false);
        }
        __syncthreads();
    }

    // epilogue: relu + fused dot with W2 chunk, reduce over 16 lanes of a row,
    // one atomic per (edge, 16-col chunk).
    const int n = lane & 15, hsel = lane >> 4;
    const float bv = b1[n0 + n];
    const float w2 = W2[n0 + n];
#pragma unroll
    for (int t = 0; t < 2; ++t) {
        v8f& acc = t ? acc1 : acc0;
#pragma unroll
        for (int r = 0; r < 8; ++r) {
            float v = fmaxf(acc[r] + bv, 0.0f) * w2;
#pragma unroll
            for (int off = 8; off > 0; off >>= 1)
                v += __shfl_xor(v, off, 16);
            if (n == 0)
                atomicAdd(&wsum[e0 + t * 16 + r + 8 * hsel], v);
        }
    }
}

// ---------------------------------------------------------------------------
// helpers
// ---------------------------------------------------------------------------

// convert + swizzle weights into WMMA B-fragment order: dst[K/32][Ncols][32]
__global__ void __launch_bounds__(256)
pack_w_k(const float* __restrict__ src, __bf16* __restrict__ dst, int K, int Ncols)
{
    int i = blockIdx.x * 256 + threadIdx.x;
    if (i < K * Ncols) {
        int k = i / Ncols, n = i - k * Ncols;
        dst[((size_t)(k >> 5) * Ncols + n) * 32 + (k & 31)] = f2bf(src[i]);
    }
}

__global__ void __launch_bounds__(256)
sigmoid_k(const float* __restrict__ wsum, const float* __restrict__ b2,
          float* __restrict__ w, int n)
{
    int i = blockIdx.x * 256 + threadIdx.x;
    if (i < n) {
        float t = wsum[i] + b2[0];
        w[i] = 1.0f / (1.0f + __expf(-t));
    }
}

// S[row[e], f] += w[e] * X[col[e], f]   (scatter-add SpMM, X bf16, S fp32)
// 2 packed bf16 features per thread; dim is a power of two (dimLog2).
__global__ void __launch_bounds__(256)
spmm_scatter(const int* __restrict__ erow, const int* __restrict__ ecol,
             const float* __restrict__ w,
             const __bf16* __restrict__ X, int ldx,
             float* __restrict__ S, int lds, int dimLog2, long long total2)
{
    long long i = (long long)blockIdx.x * 256 + threadIdx.x;
    if (i >= total2) return;
    int e  = (int)(i >> (dimLog2 - 1));
    int fp = ((int)i & ((1 << (dimLog2 - 1)) - 1)) * 2;
    unsigned xv = *(const unsigned*)(const void*)(X + (size_t)ecol[e] * ldx + fp);
    float we = w[e];
    float* sp = &S[(size_t)erow[e] * lds + fp];
    atomicAdd(sp,     we * bf2f_u16((unsigned short)(xv & 0xFFFFu)));
    atomicAdd(sp + 1, we * bf2f_u16((unsigned short)(xv >> 16)));
}

__global__ void copy_bits_k(const int* __restrict__ src, float* __restrict__ dst, int n)
{
    int i = blockIdx.x * 256 + threadIdx.x;
    if (i < n) ((int*)dst)[i] = src[i];
}

// ---------------------------------------------------------------------------
// host launcher
// ---------------------------------------------------------------------------
extern "C" void kernel_launch(void* const* d_in, const int* in_sizes, int n_in,
                              void* d_out, int out_size, void* d_ws, size_t ws_size,
                              hipStream_t stream)
{
    const float* x      = (const float*)d_in[0];
    const int*   ei     = (const int*)  d_in[1];
    const int*   gsz    = (const int*)  d_in[2];
    const float* emb_W  = (const float*)d_in[3];
    const float* emb_b  = (const float*)d_in[4];
    const float* wc1_W1 = (const float*)d_in[5];
    const float* wc1_b1 = (const float*)d_in[6];
    const float* wc1_W2 = (const float*)d_in[7];
    const float* wc1_b2 = (const float*)d_in[8];
    const float* gc1_W  = (const float*)d_in[9];
    const float* gc1_b  = (const float*)d_in[10];
    const float* wc2_W1 = (const float*)d_in[11];
    const float* wc2_b1 = (const float*)d_in[12];
    const float* wc2_W2 = (const float*)d_in[13];
    const float* wc2_b2 = (const float*)d_in[14];
    const float* gc2_W  = (const float*)d_in[15];
    const float* gc2_b  = (const float*)d_in[16];
    const float* wcl_W1 = (const float*)d_in[17];
    const float* wcl_b1 = (const float*)d_in[18];
    const float* wcl_W2 = (const float*)d_in[19];
    const float* wcl_b2 = (const float*)d_in[20];
    const float* fc_W   = (const float*)d_in[21];
    const float* fc_b   = (const float*)d_in[22];

    const int* erow = ei;
    const int* ecol = ei + GNN_E;

    // workspace layout
    char* p = (char*)d_ws;
    __bf16* xcat = (__bf16*)p;  p += (size_t)GNN_N * 768 * 2;   // [N,768] bf16 concat
    float*  S    = (float*)p;   p += (size_t)GNN_N * 512 * 4;   // [N,512] spmm accum
    float*  tmpC = (float*)p;   p += (size_t)GNN_N * 256 * 4;   // fp32 partial C
    float*  wsum = (float*)p;   p += (size_t)GNN_E * 4;
    float*  wbuf = (float*)p;   p += (size_t)GNN_E * 4;
    __bf16* emb_Wb = (__bf16*)p; p += (size_t)GNN_IN * GNN_HID * 2;
    __bf16* wc1_Wb = (__bf16*)p; p += (size_t)256 * GNN_HID * 2;
    __bf16* gc1_Wb = (__bf16*)p; p += (size_t)512 * GNN_HID * 2;
    __bf16* wc2_Wb = (__bf16*)p; p += (size_t)512 * GNN_HID * 2;
    __bf16* gc2_Wb = (__bf16*)p; p += (size_t)1024 * GNN_HID * 2;
    __bf16* wcl_Wb = (__bf16*)p; p += (size_t)768 * GNN_HID * 2;
    __bf16* fc_Wb  = (__bf16*)p; p += (size_t)768 * GNN_OUT * 2;

    float* out = (float*)d_out;                 // [N,128] ++ wl[E] ++ g_size[64]
    float* wl  = out + (size_t)GNN_N * GNN_OUT;

    const dim3 blk(256);
    const int  gM32 = (GNN_N + 31) / 32;
    const dim3 gN_2(gM32, 2), gN_1(gM32, 1);
    const dim3 gE_2(GNN_E / 32, 2);
    const int  eBlocks = (GNN_E + 255) / 256;

    // 0) per-launch weight convert + swizzle into fragment order
    auto pack = [&](const float* s, __bf16* d, int K, int Nc) {
        pack_w_k<<<(K * Nc + 255) / 256, blk, 0, stream>>>(s, d, K, Nc);
    };
    pack(emb_W,  emb_Wb, GNN_IN, GNN_HID);
    pack(wc1_W1, wc1_Wb, 256,  GNN_HID);
    pack(gc1_W,  gc1_Wb, 512,  GNN_HID);
    pack(wc2_W1, wc2_Wb, 512,  GNN_HID);
    pack(gc2_W,  gc2_Wb, 1024, GNN_HID);
    pack(wcl_W1, wcl_Wb, 768,  GNN_HID);
    pack(fc_W,   fc_Wb,  768,  GNN_OUT);

    // 1) embedding: xcat[:,0:256] = relu(x @ emb_W + b)   (A fp32)
    node_gemm<true, F_BIAS | F_RELU | F_OUTBF><<<gN_2, blk, 0, stream>>>(
        x, GNN_IN, emb_Wb, GNN_HID, emb_b, nullptr, 0,
        nullptr, xcat, 768, GNN_N, GNN_IN);

    // 2) edge weights layer 1 (K=256)
    (void)hipMemsetAsync(wsum, 0, (size_t)GNN_E * 4, stream);
    edge_gemm<<<gE_2, blk, 0, stream>>>(
        xcat, 768, erow, ecol, wc1_Wb, GNN_HID, wc1_b1, wc1_W2, wsum, 256);
    sigmoid_k<<<eBlocks, blk, 0, stream>>>(wsum, wc1_b2, wbuf, GNN_E);

    // 3) spmm (dim 256) into S
    (void)hipMemsetAsync(S, 0, (size_t)GNN_N * 512 * 4, stream);
    {
        long long total2 = (long long)GNN_E * 256 / 2;
        spmm_scatter<<<(int)((total2 + 255) / 256), blk, 0, stream>>>(
            erow, ecol, wbuf, xcat, 768, S, 512, 8, total2);
    }

    // 4) gc1: xcat[:,256:512] = relu([x1|S] @ gc1_W + b)  (two accumulating GEMMs)
    node_gemm<false, F_BIAS><<<gN_2, blk, 0, stream>>>(
        xcat, 768, gc1_Wb, GNN_HID, gc1_b, nullptr, 0,
        tmpC, nullptr, 256, GNN_N, 256);
    node_gemm<true, F_ACC | F_RELU | F_OUTBF><<<gN_2, blk, 0, stream>>>(
        S, 512, gc1_Wb + (size_t)256 * GNN_HID, GNN_HID, nullptr, tmpC, 256,
        nullptr, xcat + 256, 768, GNN_N, 256);

    // 5) edge weights layer 2 (K=512)
    (void)hipMemsetAsync(wsum, 0, (size_t)GNN_E * 4, stream);
    edge_gemm<<<gE_2, blk, 0, stream>>>(
        xcat, 768, erow, ecol, wc2_Wb, GNN_HID, wc2_b1, wc2_W2, wsum, 512);
    sigmoid_k<<<eBlocks, blk, 0, stream>>>(wsum, wc2_b2, wbuf, GNN_E);

    // 6) spmm (dim 512) into S
    (void)hipMemsetAsync(S, 0, (size_t)GNN_N * 512 * 4, stream);
    {
        long long total2 = (long long)GNN_E * 512 / 2;
        spmm_scatter<<<(int)((total2 + 255) / 256), blk, 0, stream>>>(
            erow, ecol, wbuf, xcat, 768, S, 512, 9, total2);
    }

    // 7) gc2: xcat[:,512:768] = relu([x2|S] @ gc2_W + b)
    node_gemm<false, F_BIAS><<<gN_2, blk, 0, stream>>>(
        xcat, 768, gc2_Wb, GNN_HID, gc2_b, nullptr, 0,
        tmpC, nullptr, 256, GNN_N, 512);
    node_gemm<true, F_ACC | F_RELU | F_OUTBF><<<gN_2, blk, 0, stream>>>(
        S, 512, gc2_Wb + (size_t)512 * GNN_HID, GNN_HID, nullptr, tmpC, 256,
        nullptr, xcat + 512, 768, GNN_N, 512);

    // 8) final edge weights wl (K=768) -> d_out tail
    (void)hipMemsetAsync(wsum, 0, (size_t)GNN_E * 4, stream);
    edge_gemm<<<gE_2, blk, 0, stream>>>(
        xcat, 768, erow, ecol, wcl_Wb, GNN_HID, wcl_b1, wcl_W2, wsum, 768);
    sigmoid_k<<<eBlocks, blk, 0, stream>>>(wsum, wcl_b2, wl, GNN_E);

    // 9) output projection: out = xcat @ fc_W + fc_b (fp32 out)
    node_gemm<false, F_BIAS><<<gN_1, blk, 0, stream>>>(
        xcat, 768, fc_Wb, GNN_OUT, fc_b, nullptr, 0,
        out, nullptr, GNN_OUT, GNN_N, 768);

    // 10) g_size passthrough (bit-copy int32)
    copy_bits_k<<<1, blk, 0, stream>>>(gsz, wl + GNN_E, GNN_B);
}
